// RelativeMultiHeadAttention_4930622456114
// MI455X (gfx1250) — matrix operand order repaired and checked by reading, not compiled, not run on hardware
//
#include <hip/hip_runtime.h>

#define T_LEN 128
#define B_SZ  8
#define E_DIM 1024
#define H_NUM 16
#define D_DIM 64

typedef __attribute__((ext_vector_type(16))) __bf16 v16bf;
typedef __attribute__((ext_vector_type(8)))  float  v8f;

union Frag {
  v16bf bf;
  unsigned short u[16];
  uint4 q[2];
};

__device__ __forceinline__ unsigned short f2bf(float f) {
  union { float f; unsigned u; } x; x.f = f;
  unsigned u = x.u;
  u += 0x7FFFu + ((u >> 16) & 1u);          // round-to-nearest-even
  return (unsigned short)(u >> 16);
}
__device__ __forceinline__ float bf2f(unsigned short s) {
  union { unsigned u; float f; } x; x.u = ((unsigned)s) << 16;
  return x.f;
}

// CDNA5 async copy: global -> LDS, 16B per lane, tracked by ASYNCcnt.
// Low 32 bits of a generic pointer to LDS are the workgroup-relative LDS
// address (flat-aperture rule), which is exactly what VDST wants.
__device__ __forceinline__ void async_load_b128(unsigned lds_off, const void* gptr) {
  unsigned long long ga = (unsigned long long)(uintptr_t)gptr;
  asm volatile("global_load_async_to_lds_b128 %0, %1, off"
               :: "v"(lds_off), "v"(ga) : "memory");
}
__device__ __forceinline__ void wait_async0() {
  asm volatile("s_wait_asynccnt 0x0" ::: "memory");
}

// float -> bf16 bulk conversion (8 elements/thread, b128 in / b128 out)
__global__ void __launch_bounds__(256)
cvt_bf16(const float* __restrict__ src, unsigned short* __restrict__ dst, int n) {
  int i = (blockIdx.x * 256 + threadIdx.x) * 8;
  if (i + 8 <= n) {
    float4 a = *(const float4*)(src + i);
    float4 b = *(const float4*)(src + i + 4);
    union { unsigned short u[8]; uint4 q; } o;
    o.u[0] = f2bf(a.x); o.u[1] = f2bf(a.y); o.u[2] = f2bf(a.z); o.u[3] = f2bf(a.w);
    o.u[4] = f2bf(b.x); o.u[5] = f2bf(b.y); o.u[6] = f2bf(b.z); o.u[7] = f2bf(b.w);
    *(uint4*)(dst + i) = o.q;
  }
}

// ---------------------------------------------------------------------------
// GEMM: C[r][c] = sum_k A[r][k] * W[c][k], M=N=K=1024, A/W pre-converted bf16.
// Block tile 128x64 (8 waves x 16 rows, 64 cols shared). K-step 32, fully
// unrolled (no backedge -> accumulators stay pinned, no phi copies).
// A/B tiles staged in LDS via global_load_async_to_lds_b128, double-buffered.
// mode 0: q-proj -> +spk_emb[0] chunk, q_h[n][t][d] bf16
// mode 1: k-proj -> k_h[n][t][d] bf16
// mode 2: v-proj -> v_h[n][d][t] bf16 (transposed)
// mode 3: output proj -> fp32 d_out
// ---------------------------------------------------------------------------
__global__ void __launch_bounds__(256)
gemm_wmma(const unsigned short* __restrict__ A, const unsigned short* __restrict__ W,
          const float* __restrict__ spk, unsigned short* __restrict__ dst16,
          float* __restrict__ dst32, int mode)
{
  const int K = E_DIM;
  __shared__ unsigned short lA[2][128][40];   // 32 cols used + 8 pad (80B rows)
  __shared__ unsigned short lB[2][64][40];

  const int tid = threadIdx.x;
  const int lane = tid & 31, wave = tid >> 5;
  const int half = lane >> 4, col16 = lane & 15;
  const int rowBase = blockIdx.x * 128;
  const int colBase = blockIdx.y * 64;

  // staging assignment: 16B chunks. A: 512 chunks (2/thread), B: 256 (1/thread)
  const int ar = tid >> 2, c4 = tid & 3;

  // hoisted staging addresses (2 buffers x 3 chunks) + global bases
  unsigned dA0[2], dA1[2], dB[2];
  dA0[0] = (unsigned)(uintptr_t)&lA[0][ar][c4 * 8];
  dA0[1] = (unsigned)(uintptr_t)&lA[1][ar][c4 * 8];
  dA1[0] = (unsigned)(uintptr_t)&lA[0][ar + 64][c4 * 8];
  dA1[1] = (unsigned)(uintptr_t)&lA[1][ar + 64][c4 * 8];
  dB[0]  = (unsigned)(uintptr_t)&lB[0][ar][c4 * 8];
  dB[1]  = (unsigned)(uintptr_t)&lB[1][ar][c4 * 8];
  const unsigned short* gA0 = A + (size_t)(rowBase + ar) * K + c4 * 8;
  const unsigned short* gA1 = A + (size_t)(rowBase + ar + 64) * K + c4 * 8;
  const unsigned short* gB  = W + (size_t)(colBase + ar) * K + c4 * 8;

  const v8f vzero = {0.f,0.f,0.f,0.f,0.f,0.f,0.f,0.f};
  v8f acc0 = vzero, acc1 = vzero, acc2 = vzero, acc3 = vzero;

  async_load_b128(dA0[0], gA0);
  async_load_b128(dA1[0], gA1);
  async_load_b128(dB[0],  gB);

  const int S = K / 32;                      // 32, compile-time
  #pragma unroll
  for (int s = 0; s < S; ++s) {
    wait_async0();          // this wave's stage-s loads complete
    __syncthreads();        // all waves' stage-s data visible; prev compute done
    if (s + 1 < S) {
      const int nb = (s + 1) & 1, k0 = (s + 1) * 32;
      async_load_b128(dA0[nb], gA0 + k0);
      async_load_b128(dA1[nb], gA1 + k0);
      async_load_b128(dB[nb],  gB + k0);
    }

    const int buf = s & 1;
    Frag fa;
    const unsigned short* arow = &lA[buf][wave * 16 + col16][0];
    fa.q[0] = *(const uint4*)(arow + half * 16);
    fa.q[1] = *(const uint4*)(arow + half * 16 + 8);

    Frag fb;
    const unsigned short* brow = &lB[buf][col16][half * 16];
    fb.q[0] = *(const uint4*)(brow);
    fb.q[1] = *(const uint4*)(brow + 8);
    acc0 = __builtin_amdgcn_wmma_f32_16x16x32_bf16(
        false, fa.bf, false, fb.bf, (short)0, acc0, false, false);

    brow = &lB[buf][16 + col16][half * 16];
    fb.q[0] = *(const uint4*)(brow);
    fb.q[1] = *(const uint4*)(brow + 8);
    acc1 = __builtin_amdgcn_wmma_f32_16x16x32_bf16(
        false, fa.bf, false, fb.bf, (short)0, acc1, false, false);

    brow = &lB[buf][32 + col16][half * 16];
    fb.q[0] = *(const uint4*)(brow);
    fb.q[1] = *(const uint4*)(brow + 8);
    acc2 = __builtin_amdgcn_wmma_f32_16x16x32_bf16(
        false, fa.bf, false, fb.bf, (short)0, acc2, false, false);

    brow = &lB[buf][48 + col16][half * 16];
    fb.q[0] = *(const uint4*)(brow);
    fb.q[1] = *(const uint4*)(brow + 8);
    acc3 = __builtin_amdgcn_wmma_f32_16x16x32_bf16(
        false, fa.bf, false, fb.bf, (short)0, acc3, false, false);
  }

  v8f accs[4] = {acc0, acc1, acc2, acc3};
  #pragma unroll
  for (int nt = 0; nt < 4; ++nt) {
    #pragma unroll
    for (int g = 0; g < 8; ++g) {
      int r = rowBase + wave * 16 + half * 8 + g;   // row = t*B + b
      int c = colBase + nt * 16 + col16;            // col = h*D + d
      float v = accs[nt][g];
      if (mode == 3) {
        dst32[(size_t)r * E_DIM + c] = v;
      } else {
        int t = r >> 3, b = r & 7;
        int h = c >> 6, d = c & 63;
        if (mode == 0) {
          v += spk[h * 64 + d];                     // fold s_q into new_q
          dst16[(size_t)((b * H_NUM + h) * T_LEN + t) * D_DIM + d] = f2bf(v);
        } else if (mode == 1) {
          dst16[(size_t)((b * H_NUM + h) * T_LEN + t) * D_DIM + d] = f2bf(v);
        } else {
          dst16[(size_t)((b * H_NUM + h) * D_DIM + d) * T_LEN + t] = f2bf(v);
        }
      }
    }
  }
}

// u[h][e] = sum_d spk_emb[0][h*64+d] * r_w[h*64+d][e]
__global__ void build_u(const float* __restrict__ spk, const float* __restrict__ r_w,
                        float* __restrict__ u)
{
  int h = blockIdx.x;
  for (int e = threadIdx.x; e < E_DIM; e += blockDim.x) {
    float s = 0.f;
    for (int d = 0; d < 64; ++d)
      s += spk[h * 64 + d] * r_w[(size_t)(h * 64 + d) * E_DIM + e];
    u[h * E_DIM + e] = s;
  }
}

// c[n][j] = rel_pe[j,b,:] . u[h,:]   (n = b*H + h)
__global__ void build_c(const float* __restrict__ rel_pe, const float* __restrict__ u,
                        float* __restrict__ c)
{
  int n = blockIdx.x, b = n / H_NUM, h = n % H_NUM;
  int j = threadIdx.x;
  const float* rp = rel_pe + (size_t)j * (B_SZ * E_DIM) + (size_t)b * E_DIM;
  const float* uh = u + h * E_DIM;
  float s = 0.f;
  for (int e = 0; e < E_DIM; ++e) s += rp[e] * uh[e];
  c[n * T_LEN + j] = s;
}

// ---------------------------------------------------------------------------
// Attention per head n: scores (WMMA) + a2/a3 + faithful 1e-30 mask softmax
// + probs@V (WMMA). One block = one head, 8 waves x 16 rows each.
// ---------------------------------------------------------------------------
__global__ void __launch_bounds__(256)
attn_wmma(const unsigned short* __restrict__ qh, const unsigned short* __restrict__ kh,
          const unsigned short* __restrict__ vh, const float* __restrict__ cvec,
          const int* __restrict__ spkm, const float* __restrict__ spk_emb,
          unsigned short* __restrict__ attn_out)
{
  __shared__ float lds_c[T_LEN];
  __shared__ float lds_d0[T_LEN];
  __shared__ float lds_d1[T_LEN];
  __shared__ unsigned short lds_w[8][16][136];   // padded rows: bank-spread

  const int n = blockIdx.x, b = n / H_NUM, h = n % H_NUM;
  const int tid = threadIdx.x;
  const int lane = tid & 31, wave = tid >> 5;
  const int half = lane >> 4, col16 = lane & 15;

  if (tid < T_LEN) lds_c[tid] = cvec[n * T_LEN + tid];
  {
    int i = tid & 127, which = tid >> 7;
    const unsigned short* qr = qh + ((size_t)n * T_LEN + i) * D_DIM;
    const float* er = spk_emb + which * E_DIM + h * 64;
    float s = 0.f;
    for (int d = 0; d < 64; ++d) s += bf2f(qr[d]) * er[d];
    if (which == 0) lds_d0[i] = s; else lds_d1[i] = s;
  }
  __syncthreads();

  const int iBase = wave * 16;
  const v8f vzero = {0.f,0.f,0.f,0.f,0.f,0.f,0.f,0.f};

  Frag fa[2];
  {
    const unsigned short* qr = qh + ((size_t)n * T_LEN + iBase + col16) * D_DIM;
    #pragma unroll
    for (int ks = 0; ks < 2; ++ks) {
      fa[ks].q[0] = *(const uint4*)(qr + ks * 32 + half * 16);
      fa[ks].q[1] = *(const uint4*)(qr + ks * 32 + half * 16 + 8);
    }
  }

  v8f sacc[8];
  #pragma unroll
  for (int jt = 0; jt < 8; ++jt) sacc[jt] = vzero;
  #pragma unroll
  for (int jt = 0; jt < 8; ++jt) {
    #pragma unroll
    for (int ks = 0; ks < 2; ++ks) {
      const unsigned short* kr =
          kh + ((size_t)n * T_LEN + jt * 16 + col16) * D_DIM + ks * 32 + half * 16;
      Frag fb;
      fb.q[0] = *(const uint4*)(kr);
      fb.q[1] = *(const uint4*)(kr + 8);
      sacc[jt] = __builtin_amdgcn_wmma_f32_16x16x32_bf16(
          false, fa[ks].bf, false, fb.bf, (short)0, sacc[jt], false, false);
    }
  }

  #pragma unroll
  for (int g = 0; g < 8; ++g) {
    int M = half * 8 + g;
    int i = iBase + M;
    float d0v = lds_d0[i], d1v = lds_d1[i];
    float lg[8];
    #pragma unroll
    for (int jt = 0; jt < 8; ++jt) {
      int j = jt * 16 + col16;
      float s = sacc[jt][g];
      int sm = spkm[((size_t)b * T_LEN + i) * T_LEN + j];
      float l;
      if (j <= i) l = (s + (sm ? d1v : d0v) + lds_c[127 - i + j]) * 0.125f;
      else        l = 1e-30f;                      // faithful mask value
      lg[jt] = l;
    }
    float mx = lg[0];
    #pragma unroll
    for (int jt = 1; jt < 8; ++jt) mx = fmaxf(mx, lg[jt]);
    #pragma unroll
    for (int m = 1; m < 16; m <<= 1) mx = fmaxf(mx, __shfl_xor(mx, m, 32));
    float sum = 0.f;
    #pragma unroll
    for (int jt = 0; jt < 8; ++jt) { lg[jt] = __expf(lg[jt] - mx); sum += lg[jt]; }
    #pragma unroll
    for (int m = 1; m < 16; m <<= 1) sum += __shfl_xor(sum, m, 32);
    float inv = 1.0f / sum;
    #pragma unroll
    for (int jt = 0; jt < 8; ++jt) {
      int j = jt * 16 + col16;
      lds_w[wave][M][j] = f2bf(lg[jt] * inv);
    }
  }
  // same-wave LDS RAW is in-order; compiler inserts s_wait_dscnt as needed.

  v8f oacc[4];
  #pragma unroll
  for (int nt = 0; nt < 4; ++nt) oacc[nt] = vzero;
  #pragma unroll
  for (int jk = 0; jk < 128; jk += 32) {
    Frag fw;
    const unsigned short* wr = &lds_w[wave][col16][0];
    fw.q[0] = *(const uint4*)(wr + jk + half * 16);
    fw.q[1] = *(const uint4*)(wr + jk + half * 16 + 8);
    #pragma unroll
    for (int nt = 0; nt < 4; ++nt) {
      const unsigned short* vr =
          vh + ((size_t)n * D_DIM + nt * 16 + col16) * T_LEN + jk + half * 16;
      Frag fv;
      fv.q[0] = *(const uint4*)(vr);
      fv.q[1] = *(const uint4*)(vr + 8);
      oacc[nt] = __builtin_amdgcn_wmma_f32_16x16x32_bf16(
          false, fw.bf, false, fv.bf, (short)0, oacc[nt], false, false);
    }
  }

  #pragma unroll
  for (int nt = 0; nt < 4; ++nt) {
    #pragma unroll
    for (int g = 0; g < 8; ++g) {
      int t = iBase + half * 8 + g;
      int d = nt * 16 + col16;
      attn_out[((size_t)t * B_SZ + b) * E_DIM + h * 64 + d] = f2bf(oacc[nt][g]);
    }
  }
}

extern "C" void kernel_launch(void* const* d_in, const int* in_sizes, int n_in,
                              void* d_out, int out_size, void* d_ws, size_t ws_size,
                              hipStream_t stream) {
  (void)in_sizes; (void)n_in; (void)out_size; (void)ws_size;
  const float* query  = (const float*)d_in[0];
  const float* rel_pe = (const float*)d_in[3];
  const int*   spkm   = (const int*)d_in[6];
  const float* q_w    = (const float*)d_in[7];
  const float* k_w    = (const float*)d_in[8];
  const float* v_w    = (const float*)d_in[9];
  const float* r_w    = (const float*)d_in[10];
  const float* spk    = (const float*)d_in[11];
  const float* out_w  = (const float*)d_in[12];
  float* out = (float*)d_out;

  char* ws = (char*)d_ws;
  unsigned short* q_h  = (unsigned short*)(ws);                  // 2 MB
  unsigned short* k_h  = (unsigned short*)(ws + (2ull << 20));   // 2 MB
  unsigned short* v_h  = (unsigned short*)(ws + (4ull << 20));   // 2 MB
  unsigned short* attn = (unsigned short*)(ws + (6ull << 20));   // 2 MB
  unsigned short* qbf  = (unsigned short*)(ws + (8ull << 20));   // 2 MB
  unsigned short* wbf  = (unsigned short*)(ws + (10ull << 20));  // 2 MB (reused)
  float* u_buf = (float*)(ws + (12ull << 20));                   // 64 KB
  float* c_buf = (float*)(ws + (12ull << 20) + (1ull << 16));    // 64 KB

  const int NE = E_DIM * E_DIM;                 // 1M elements
  dim3 cg(NE / (256 * 8)), cb(256);
  dim3 gg(8, 16), gb(256);

  cvt_bf16<<<cg, cb, 0, stream>>>(query, qbf, NE);

  cvt_bf16<<<cg, cb, 0, stream>>>(q_w, wbf, NE);
  gemm_wmma<<<gg, gb, 0, stream>>>(qbf, wbf, spk, q_h, nullptr, 0);
  cvt_bf16<<<cg, cb, 0, stream>>>(k_w, wbf, NE);
  gemm_wmma<<<gg, gb, 0, stream>>>(qbf, wbf, spk, k_h, nullptr, 1);
  cvt_bf16<<<cg, cb, 0, stream>>>(v_w, wbf, NE);
  gemm_wmma<<<gg, gb, 0, stream>>>(qbf, wbf, spk, v_h, nullptr, 2);

  build_u<<<dim3(H_NUM), dim3(256), 0, stream>>>(spk, r_w, u_buf);
  build_c<<<dim3(B_SZ * H_NUM), dim3(T_LEN), 0, stream>>>(rel_pe, u_buf, c_buf);
  attn_wmma<<<dim3(B_SZ * H_NUM), dim3(256), 0, stream>>>(q_h, k_h, v_h, c_buf,
                                                          spkm, spk, attn);

  cvt_bf16<<<cg, cb, 0, stream>>>(out_w, wbf, NE);
  gemm_wmma<<<gg, gb, 0, stream>>>(attn, wbf, spk, nullptr, out, 3);
}
